// Decoder_LSTM_with_attention_60481729462798
// MI455X (gfx1250) — matrix operand order repaired and checked by reading, not compiled, not run on hardware
//
#include <hip/hip_runtime.h>
#include <hip/hip_bf16.h>

// ---------------------------------------------------------------------------
// Decoder LSTM with additive attention, greedy autoregressive decode.
// B=32, S=120, H=512, E=200, V=5000, 64 steps.
//
// Strategy (MI455X / gfx1250):
//  * All per-step GEMMs are M=32 weight-streaming GEMMs. fp32 weights
//    (232 MB/step, W2 alone = 200 MB > 192 MB L2) would re-stream from HBM
//    every step (~14.8 GB total). We convert all weights to bf16 once per
//    launch (~116 MB total -> L2 resident), WMMA bf16 with f32 accumulate.
//  * Weights are pre-swizzled into the exact V_WMMA_F32_16X16X32_BF16
//    B-fragment layout so the inner loop loads one contiguous 32B vector
//    per lane.
//  * Each wave computes a 32(M) x 16(N) output slab (2 WMMAs per K-chunk,
//    B fragment reused across both M tiles).
//  * fc2 has only 313 N-tiles -> split-K across 8 waves/block + LDS reduce.
// ---------------------------------------------------------------------------

typedef __attribute__((ext_vector_type(16))) __bf16 v16bf;
typedef __attribute__((ext_vector_type(8)))  __bf16 v8bf;
typedef __attribute__((ext_vector_type(8)))  float  v8f;

#define B_     32
#define S_     120
#define H_     512
#define E_     200
#define V_     5000
#define D_     712          // E + H (W_ih input size)
#define G_     2048         // 4*H
#define KCAT   1248         // D + H padded to mult of 32
#define KC_CAT 39           // KCAT/32
#define NT_CAT 128          // G/16
#define N1_    10000        // 2V
#define NT1    625          // N1/16
#define KC1    16           // 512/32
#define HPAD   10016        // N1 padded to mult of 32 (fc2 K)
#define KC2    313          // HPAD/32
#define NT2    313          // 5008/16
#define TMAX   64
#define KEYR   3840         // B*S
#define KEYMT  240          // KEYR/16
#define KEYNT  32           // H/16

static __device__ __forceinline__ v8f wmma_bf16(v16bf a, v16bf b, v8f c) {
  return __builtin_amdgcn_wmma_f32_16x16x32_bf16(
      /*neg_a=*/false, a, /*neg_b=*/false, b,
      /*c_mod=*/(short)0, c, /*reuse_a=*/false, /*reuse_b=*/false);
}

// A fragment (16x32 bf16, row-major source A[row][k], lda in elements).
// Lane L<16:  row=m0+L,    elems 0..7 = K k0..k0+7,  8..15 = K k0+16..k0+23
// Lane L>=16: row=m0+L-16, elems 0..7 = K k0+8..+15, 8..15 = K k0+24..+31
static __device__ __forceinline__ v16bf load_a_frag(const __bf16* A, int lda,
                                                    int m0, int k0, int lane) {
  int row = m0 + (lane & 15);
  int kb  = k0 + ((lane & 16) >> 1);   // +8 for upper half-wave
  const __bf16* p = A + (size_t)row * lda + kb;
  v8bf lo = *(const v8bf*)(p);
  v8bf hi = *(const v8bf*)(p + 16);
  return __builtin_shufflevector(lo, hi, 0, 1, 2, 3, 4, 5, 6, 7,
                                 8, 9, 10, 11, 12, 13, 14, 15);
}

// ---------------------------------------------------------------------------
// One-time setup kernels
// ---------------------------------------------------------------------------

__global__ void k_init(const float* __restrict__ h0, const float* __restrict__ c0,
                       float* __restrict__ h, float* __restrict__ c,
                       int* __restrict__ dec, __bf16* __restrict__ hidden) {
  int i = blockIdx.x * blockDim.x + threadIdx.x;
  if (i < B_ * H_) { h[i] = h0[i]; c[i] = c0[i]; }
  if (i < B_) dec[i] = 0;                       // START_TOKEN
  for (int j = i; j < B_ * HPAD; j += gridDim.x * blockDim.x)
    hidden[j] = (__bf16)0.f;                    // keep fc2 K-pad columns zero
}

__global__ void k_f32_to_bf16(const float* __restrict__ src,
                              __bf16* __restrict__ dst, int n) {
  for (int i = blockIdx.x * blockDim.x + threadIdx.x; i < n;
       i += gridDim.x * blockDim.x)
    dst[i] = (__bf16)src[i];
}

// Swizzle W (row-major [N][K] fp32, zero-padded to tiles) into WMMA B-fragment
// order: frag[((tile*nKc + kc)*32 + lane)*16 + e]
//   lane L<16:  N = tile*16+L,    e 0..7 = K kc*32+e,   8..15 = K kc*32+16+(e-8)
//   lane L>=16: N = tile*16+L-16, e 0..7 = K kc*32+8+e, 8..15 = K kc*32+24+(e-8)
__global__ void k_build_frag(const float* __restrict__ W, int N, int K,
                             __bf16* __restrict__ out, int nTiles, int nKc) {
  int t = blockIdx.x * blockDim.x + threadIdx.x;
  int total = nTiles * nKc * 32;
  if (t >= total) return;
  int lane = t & 31;
  int idx  = t >> 5;
  int kc   = idx % nKc;
  int tile = idx / nKc;
  int n  = tile * 16 + (lane & 15);
  int ks = kc * 32 + ((lane & 16) >> 1);
  __bf16* o = out + (size_t)t * 16;
#pragma unroll
  for (int e = 0; e < 16; ++e) {
    int k = ks + (e & 7) + ((e & 8) << 1);
    float v = (n < N && k < K) ? W[(size_t)n * K + k] : 0.f;
    o[e] = (__bf16)v;
  }
}

// Fused [W_ih | W_hh | 0] fragment builder: N=2048, K: [0,712)=W_ih,
// [712,1224)=W_hh, [1224,1248)=0.
__global__ void k_build_frag_cat(const float* __restrict__ Wih,
                                 const float* __restrict__ Whh,
                                 __bf16* __restrict__ out) {
  int t = blockIdx.x * blockDim.x + threadIdx.x;
  int total = NT_CAT * KC_CAT * 32;
  if (t >= total) return;
  int lane = t & 31;
  int idx  = t >> 5;
  int kc   = idx % KC_CAT;
  int tile = idx / KC_CAT;
  int n  = tile * 16 + (lane & 15);
  int ks = kc * 32 + ((lane & 16) >> 1);
  __bf16* o = out + (size_t)t * 16;
#pragma unroll
  for (int e = 0; e < 16; ++e) {
    int k = ks + (e & 7) + ((e & 8) << 1);
    float v;
    if (k < D_)            v = Wih[(size_t)n * D_ + k];
    else if (k < D_ + H_)  v = Whh[(size_t)n * H_ + (k - D_)];
    else                   v = 0.f;
    o[e] = (__bf16)v;
  }
}

// keys = enc @ Wk^T + bk   (one-time). M=3840, N=512, K=512, WMMA bf16.
__global__ void k_gemm_keys(const __bf16* __restrict__ A,     // encb [3840][512]
                            const __bf16* __restrict__ Bfrag, // Wk frag
                            const float* __restrict__ bk,
                            float* __restrict__ keys) {
  int wave = threadIdx.x >> 5, lane = threadIdx.x & 31;
  int gw = blockIdx.x * (blockDim.x >> 5) + wave;
  int mt = gw >> 5;        // KEYNT = 32 N-tiles
  int nt = gw & 31;
  if (mt >= KEYMT) return;
  v8f acc = {};
  const v16bf* bp = (const v16bf*)Bfrag + (size_t)nt * KC1 * 32 + lane;
  const __bf16* At = A + (size_t)mt * 16 * H_;
#pragma unroll 4
  for (int kc = 0; kc < KC1; ++kc) {
    v16bf b = bp[(size_t)kc * 32];
    v16bf a = load_a_frag(At, H_, 0, kc * 32, lane);
    acc = wmma_bf16(a, b, acc);
  }
  int n = nt * 16 + (lane & 15);
  int off = (lane & 16) ? 8 : 0;
  float bkn = bk[n];
#pragma unroll
  for (int i = 0; i < 8; ++i)
    keys[(size_t)(mt * 16 + off + i) * H_ + n] = acc[i] + bkn;
}

// ---------------------------------------------------------------------------
// Per-step kernels
// ---------------------------------------------------------------------------

// Additive attention + build bf16 LSTM input x_cat = [emb | context | h | 0].
__global__ void k_attn(const float* __restrict__ h, const float* __restrict__ keys,
                       const float* __restrict__ enc, const float* __restrict__ Wq,
                       const float* __restrict__ bq, const float* __restrict__ Wv,
                       const float* __restrict__ bv, const float* __restrict__ emb,
                       const int* __restrict__ dec, __bf16* __restrict__ xcat) {
  __shared__ float hb[H_];
  __shared__ float q[H_];
  __shared__ float sc[S_];
  int b = blockIdx.x, tid = threadIdx.x;          // 256 threads
  for (int j = tid; j < H_; j += 256) hb[j] = h[(size_t)b * H_ + j];
  __syncthreads();
  for (int j = tid; j < H_; j += 256) {           // q = h @ Wq^T + bq
    const float* wr = Wq + (size_t)j * H_;
    float a = bq[j];
    for (int k = 0; k < H_; ++k) a = fmaf(wr[k], hb[k], a);
    q[j] = a;
  }
  __syncthreads();
  for (int s = tid; s < S_; s += 256) {           // scores
    const float* kr = keys + ((size_t)b * S_ + s) * H_;
    float a = bv[0];
    for (int k = 0; k < H_; ++k) a = fmaf(tanhf(q[k] + kr[k]), Wv[k], a);
    sc[s] = a;
  }
  __syncthreads();
  if (tid == 0) {                                 // softmax over 120 (tiny)
    float mx = -1e30f;
    for (int s = 0; s < S_; ++s) mx = fmaxf(mx, sc[s]);
    float sum = 0.f;
    for (int s = 0; s < S_; ++s) { float e = __expf(sc[s] - mx); sc[s] = e; sum += e; }
    float inv = 1.f / sum;
    for (int s = 0; s < S_; ++s) sc[s] *= inv;
  }
  __syncthreads();
  __bf16* xr = xcat + (size_t)b * KCAT;
  for (int j = tid; j < H_; j += 256) {           // context + copy h
    float a = 0.f;
    for (int s = 0; s < S_; ++s)
      a = fmaf(sc[s], enc[((size_t)b * S_ + s) * H_ + j], a);
    xr[E_ + j]      = (__bf16)a;
    xr[E_ + H_ + j] = (__bf16)hb[j];
  }
  int tok = dec[b];
  for (int j = tid; j < E_; j += 256) xr[j] = (__bf16)emb[(size_t)tok * E_ + j];
  for (int j = tid; j < KCAT - E_ - 2 * H_; j += 256)
    xr[E_ + 2 * H_ + j] = (__bf16)0.f;            // K pad
}

// gates[32,2048] = x_cat @ [W_ih|W_hh]^T  (biases added in pointwise kernel)
__global__ void k_gemm_gates(const __bf16* __restrict__ xcat,
                             const __bf16* __restrict__ Bfrag,
                             float* __restrict__ gates) {
  int wave = threadIdx.x >> 5, lane = threadIdx.x & 31;
  int nt = blockIdx.x * (blockDim.x >> 5) + wave;
  if (nt >= NT_CAT) return;
  v8f acc0 = {}, acc1 = {};
  const v16bf* bp = (const v16bf*)Bfrag + (size_t)nt * KC_CAT * 32 + lane;
#pragma unroll 2
  for (int kc = 0; kc < KC_CAT; ++kc) {
    v16bf b  = bp[(size_t)kc * 32];
    v16bf a0 = load_a_frag(xcat, KCAT, 0, kc * 32, lane);
    v16bf a1 = load_a_frag(xcat, KCAT, 16, kc * 32, lane);
    acc0 = wmma_bf16(a0, b, acc0);
    acc1 = wmma_bf16(a1, b, acc1);
  }
  int n = nt * 16 + (lane & 15);
  int off = (lane & 16) ? 8 : 0;
#pragma unroll
  for (int i = 0; i < 8; ++i) {
    gates[(size_t)(off + i) * G_ + n]      = acc0[i];
    gates[(size_t)(16 + off + i) * G_ + n] = acc1[i];
  }
}

__global__ void k_lstm(const float* __restrict__ gates,
                       const float* __restrict__ bih, const float* __restrict__ bhh,
                       float* __restrict__ c, float* __restrict__ h,
                       __bf16* __restrict__ hbf) {
  int idx = blockIdx.x * blockDim.x + threadIdx.x;    // B*H = 16384
  if (idx >= B_ * H_) return;
  int b = idx >> 9, j = idx & (H_ - 1);
  const float* g = gates + (size_t)b * G_;
  float gi = g[j]            + bih[j]            + bhh[j];
  float gf = g[H_ + j]       + bih[H_ + j]       + bhh[H_ + j];
  float gg = g[2 * H_ + j]   + bih[2 * H_ + j]   + bhh[2 * H_ + j];
  float go = g[3 * H_ + j]   + bih[3 * H_ + j]   + bhh[3 * H_ + j];
  float si = 1.f / (1.f + __expf(-gi));
  float sf = 1.f / (1.f + __expf(-gf));
  float so = 1.f / (1.f + __expf(-go));
  float cn = sf * c[idx] + si * tanhf(gg);
  float hn = so * tanhf(cn);
  c[idx] = cn;
  h[idx] = hn;
  hbf[idx] = (__bf16)hn;
}

// hidden[32,10016] = relu(h @ W1^T + b1), stored bf16 (K-pad cols stay 0).
__global__ void k_gemm_fc1(const __bf16* __restrict__ hbf,
                           const __bf16* __restrict__ Bfrag,
                           const float* __restrict__ b1,
                           __bf16* __restrict__ hidden) {
  int wave = threadIdx.x >> 5, lane = threadIdx.x & 31;
  int nt = blockIdx.x * (blockDim.x >> 5) + wave;
  if (nt >= NT1) return;
  v8f acc0 = {}, acc1 = {};
  const v16bf* bp = (const v16bf*)Bfrag + (size_t)nt * KC1 * 32 + lane;
#pragma unroll 4
  for (int kc = 0; kc < KC1; ++kc) {
    v16bf b  = bp[(size_t)kc * 32];
    v16bf a0 = load_a_frag(hbf, H_, 0, kc * 32, lane);
    v16bf a1 = load_a_frag(hbf, H_, 16, kc * 32, lane);
    acc0 = wmma_bf16(a0, b, acc0);
    acc1 = wmma_bf16(a1, b, acc1);
  }
  int n = nt * 16 + (lane & 15);
  int off = (lane & 16) ? 8 : 0;
  float bn = b1[n];
#pragma unroll
  for (int i = 0; i < 8; ++i) {
    hidden[(size_t)(off + i) * HPAD + n]      = (__bf16)fmaxf(acc0[i] + bn, 0.f);
    hidden[(size_t)(16 + off + i) * HPAD + n] = (__bf16)fmaxf(acc1[i] + bn, 0.f);
  }
}

// logits = hidden @ W2^T + b2, written straight into d_out[b][t][v].
// Split-K over 8 waves per block (KC2=313), LDS reduction.
__global__ void k_gemm_fc2(const __bf16* __restrict__ hidden,
                           const __bf16* __restrict__ Bfrag,
                           const float* __restrict__ b2,
                           float* __restrict__ out, int t) {
  __shared__ float red[8][32][16];                 // 16 KB
  int wave = threadIdx.x >> 5, lane = threadIdx.x & 31;
  int nt = blockIdx.x;                             // 313 blocks
  v8f acc0 = {}, acc1 = {};
  const v16bf* bp = (const v16bf*)Bfrag + (size_t)nt * KC2 * 32 + lane;
  for (int kc = wave; kc < KC2; kc += 8) {
    v16bf b  = bp[(size_t)kc * 32];
    v16bf a0 = load_a_frag(hidden, HPAD, 0, kc * 32, lane);
    v16bf a1 = load_a_frag(hidden, HPAD, 16, kc * 32, lane);
    acc0 = wmma_bf16(a0, b, acc0);
    acc1 = wmma_bf16(a1, b, acc1);
  }
#pragma unroll
  for (int i = 0; i < 8; ++i) {
    red[wave][lane][i]     = acc0[i];
    red[wave][lane][8 + i] = acc1[i];
  }
  __syncthreads();
  for (int p = threadIdx.x; p < 512; p += 256) {
    int ln = p >> 4, i = p & 15;
    float s = 0.f;
#pragma unroll
    for (int w = 0; w < 8; ++w) s += red[w][ln][i];
    int n = nt * 16 + (ln & 15);
    int m = (i & 7) + ((ln >= 16) ? 8 : 0) + ((i >= 8) ? 16 : 0);
    if (n < V_)
      out[((size_t)m * TMAX + t) * V_ + n] = s + b2[n];
  }
}

__global__ void k_argmax(const float* __restrict__ out, int t,
                         int* __restrict__ dec) {
  __shared__ float sv[256];
  __shared__ int   si[256];
  int b = blockIdx.x, tid = threadIdx.x;
  const float* row = out + ((size_t)b * TMAX + t) * V_;
  float best = -1e30f; int bi = 0;
  for (int v = tid; v < V_; v += 256) {
    float x = row[v];
    if (x > best) { best = x; bi = v; }
  }
  sv[tid] = best; si[tid] = bi;
  __syncthreads();
  for (int s = 128; s > 0; s >>= 1) {
    if (tid < s) {
      if (sv[tid + s] > sv[tid] ||
          (sv[tid + s] == sv[tid] && si[tid + s] < si[tid])) {
        sv[tid] = sv[tid + s]; si[tid] = si[tid + s];
      }
    }
    __syncthreads();
  }
  if (tid == 0) dec[b] = si[0];
}

// ---------------------------------------------------------------------------
// Workspace layout
// ---------------------------------------------------------------------------
static constexpr size_t AL(size_t x) { return (x + 255) & ~(size_t)255; }
static constexpr size_t OFF_KEYS  = 0;                                           // f32 3840*512
static constexpr size_t OFF_ENCB  = AL(OFF_KEYS  + (size_t)KEYR * H_ * 4);       // bf16 3840*512
static constexpr size_t OFF_WKF   = AL(OFF_ENCB  + (size_t)KEYR * H_ * 2);       // bf16 512*512
static constexpr size_t OFF_WCF   = AL(OFF_WKF   + (size_t)H_ * H_ * 2);         // bf16 2048*1248
static constexpr size_t OFF_W1F   = AL(OFF_WCF   + (size_t)G_ * KCAT * 2);       // bf16 10000*512 (padded)
static constexpr size_t OFF_W2F   = AL(OFF_W1F   + (size_t)NT1 * KC1 * 512 * 2);
static constexpr size_t OFF_H     = AL(OFF_W2F   + (size_t)NT2 * KC2 * 512 * 2); // ~100.3 MB frag
static constexpr size_t OFF_C     = AL(OFF_H     + (size_t)B_ * H_ * 4);
static constexpr size_t OFF_HBF   = AL(OFF_C     + (size_t)B_ * H_ * 4);
static constexpr size_t OFF_XCAT  = AL(OFF_HBF   + (size_t)B_ * H_ * 2);
static constexpr size_t OFF_GATES = AL(OFF_XCAT  + (size_t)B_ * KCAT * 2);
static constexpr size_t OFF_HID   = AL(OFF_GATES + (size_t)B_ * G_ * 4);
static constexpr size_t OFF_DEC   = AL(OFF_HID   + (size_t)B_ * HPAD * 2);

extern "C" void kernel_launch(void* const* d_in, const int* in_sizes, int n_in,
                              void* d_out, int out_size, void* d_ws, size_t ws_size,
                              hipStream_t stream) {
  (void)in_sizes; (void)n_in; (void)out_size; (void)ws_size;
  const float* enc = (const float*)d_in[0];
  const float* h0  = (const float*)d_in[1];
  const float* c0  = (const float*)d_in[2];
  const float* emb = (const float*)d_in[3];
  const float* Wq  = (const float*)d_in[4];
  const float* bq  = (const float*)d_in[5];
  const float* Wk  = (const float*)d_in[6];
  const float* bk  = (const float*)d_in[7];
  const float* Wv  = (const float*)d_in[8];
  const float* bv  = (const float*)d_in[9];
  const float* Wih = (const float*)d_in[10];
  const float* Whh = (const float*)d_in[11];
  const float* bih = (const float*)d_in[12];
  const float* bhh = (const float*)d_in[13];
  const float* W1  = (const float*)d_in[14];
  const float* b1  = (const float*)d_in[15];
  const float* W2  = (const float*)d_in[16];
  const float* b2  = (const float*)d_in[17];
  float* out = (float*)d_out;

  char* ws = (char*)d_ws;
  float*  keys   = (float*)(ws + OFF_KEYS);
  __bf16* encb   = (__bf16*)(ws + OFF_ENCB);
  __bf16* wkf    = (__bf16*)(ws + OFF_WKF);
  __bf16* wcf    = (__bf16*)(ws + OFF_WCF);
  __bf16* w1f    = (__bf16*)(ws + OFF_W1F);
  __bf16* w2f    = (__bf16*)(ws + OFF_W2F);
  float*  h      = (float*)(ws + OFF_H);
  float*  c      = (float*)(ws + OFF_C);
  __bf16* hbf    = (__bf16*)(ws + OFF_HBF);
  __bf16* xcat   = (__bf16*)(ws + OFF_XCAT);
  float*  gates  = (float*)(ws + OFF_GATES);
  __bf16* hidden = (__bf16*)(ws + OFF_HID);
  int*    dec    = (int*)(ws + OFF_DEC);

  // ---- one-time per launch: state init, bf16 weight fragments, keys ----
  k_init<<<(B_ * HPAD + 255) / 256, 256, 0, stream>>>(h0, c0, h, c, dec, hidden);
  k_f32_to_bf16<<<1024, 256, 0, stream>>>(enc, encb, KEYR * H_);
  k_build_frag<<<(KEYNT * KC1 * 32 + 255) / 256, 256, 0, stream>>>(
      Wk, H_, H_, wkf, KEYNT, KC1);
  k_build_frag_cat<<<(NT_CAT * KC_CAT * 32 + 255) / 256, 256, 0, stream>>>(
      Wih, Whh, wcf);
  k_build_frag<<<(NT1 * KC1 * 32 + 255) / 256, 256, 0, stream>>>(
      W1, N1_, H_, w1f, NT1, KC1);
  k_build_frag<<<(NT2 * KC2 * 32 + 255) / 256, 256, 0, stream>>>(
      W2, V_, N1_, w2f, NT2, KC2);
  k_gemm_keys<<<(KEYMT * KEYNT + 3) / 4, 128, 0, stream>>>(encb, wkf, bk, keys);

  // ---- 64 autoregressive steps ----
  for (int t = 0; t < TMAX; ++t) {
    k_attn<<<B_, 256, 0, stream>>>(h, keys, enc, Wq, bq, Wv, bv, emb, dec, xcat);
    k_gemm_gates<<<(NT_CAT + 3) / 4, 128, 0, stream>>>(xcat, wcf, gates);
    k_lstm<<<(B_ * H_ + 255) / 256, 256, 0, stream>>>(gates, bih, bhh, c, h, hbf);
    k_gemm_fc1<<<(NT1 + 3) / 4, 128, 0, stream>>>(hbf, w1f, b1, hidden);
    k_gemm_fc2<<<NT2, 256, 0, stream>>>(hidden, w2f, b2, out, t);
    k_argmax<<<B_, 256, 0, stream>>>(out, t, dec);
  }
}
